// Decoder_9483287790289
// MI455X (gfx1250) — matrix operand order repaired and checked by reading, not compiled, not run on hardware
//
#include <hip/hip_runtime.h>
#include <hip/hip_bf16.h>

// ---------------------------------------------------------------------------
// GRU decoder for MI455X (gfx1250, wave32, WMMA).
// Heavy GEMMs run as f16 WMMA (V_WMMA_F32_16X16X32_F16) with f32 accumulate.
// Wave tile 32x64 (8 accumulators): 8 WMMAs per 12 B128 loads per K-step,
// ~21 FLOP per L2 byte (vs 12.8 at 16x64) to stay under the L2 roof.
// ---------------------------------------------------------------------------

#define BATCH   64
#define TLEN    64
#define EMB     300
#define EMB_P   320    // pad to mult of 32
#define HID     700
#define HID_P   704    // pad to mult of 32
#define GATES   2100   // 3*HID
#define GATES_P 2112   // pad to mult of 64
#define VOCAB   32000
#define ROWS    (BATCH * TLEN)   // 4096

typedef __attribute__((ext_vector_type(16))) _Float16 v16h;
typedef __attribute__((ext_vector_type(8)))  _Float16 v8h;
typedef __attribute__((ext_vector_type(8)))  float    v8f;

// ---------------------------------------------------------------------------
// h0 = concat(z, attr_emb[l]) with zero pad to HID_P; fp32 + f16 copies
// ---------------------------------------------------------------------------
__global__ void k_prep_h0(const float* __restrict__ z, const int* __restrict__ l,
                          const float* __restrict__ attr_emb,
                          float* __restrict__ h, _Float16* __restrict__ h16) {
    int idx = blockIdx.x * blockDim.x + threadIdx.x;
    if (idx >= BATCH * HID_P) return;
    int b = idx / HID_P, j = idx % HID_P;
    float v = 0.f;
    if (j < 500)       v = z[b * 500 + j];
    else if (j < HID)  v = attr_emb[l[b] * (HID - 500) + (j - 500)];
    h[idx]   = v;
    h16[idx] = (_Float16)v;
}

// ---------------------------------------------------------------------------
// fp32 (rs x cs) -> f16 (rd x cd) with zero padding
// ---------------------------------------------------------------------------
__global__ void k_cvt_pad(const float* __restrict__ src, _Float16* __restrict__ dst,
                          int rs, int cs, int rd, int cd) {
    int idx = blockIdx.x * blockDim.x + threadIdx.x;
    if (idx >= rd * cd) return;
    int r = idx / cd, c = idx % cd;
    dst[idx] = (r < rs && c < cs) ? (_Float16)src[(size_t)r * cs + c] : (_Float16)0.f;
}

// ---------------------------------------------------------------------------
// xe[i, k] = emb[x_flat[i], k] as f16, K padded 300 -> 320
// ---------------------------------------------------------------------------
__global__ void k_gather_emb(const int* __restrict__ x, const float* __restrict__ emb,
                             _Float16* __restrict__ xe) {
    int idx = blockIdx.x * blockDim.x + threadIdx.x;
    if (idx >= ROWS * EMB_P) return;
    int i = idx / EMB_P, k = idx % EMB_P;
    xe[idx] = (k < EMB) ? (_Float16)emb[(size_t)x[i] * EMB + k] : (_Float16)0.f;
}

// ---------------------------------------------------------------------------
// Generic f16 WMMA GEMM:  C(MxN,f32) = A(MxK,f16) * W(NxK,f16)^T [+ bias]
// Block = 256 threads = 8 waves arranged 2(M) x 4(N); wave tile = 32x64
// (2 M-tiles x 4 N-tiles = 8 f32 accumulators). Block tile = 64x256.
// Requires: K % 32 == 0, M % 64 == 0 (grid.y = M/64), N % 64 == 0.
// All early exits are wave-uniform -> EXEC all-1s at every WMMA.
// ---------------------------------------------------------------------------
__global__ void __launch_bounds__(256)
k_wmma_gemm(const _Float16* __restrict__ A, const _Float16* __restrict__ W,
            float* __restrict__ C, const float* __restrict__ bias,
            int N, int K, int lda, int ldb, int ldc, int nt_store) {
    const int lane = threadIdx.x & 31;
    const int wave = threadIdx.x >> 5;
    const int wmg = wave >> 2;                // 0..1  (M group)
    const int wng = wave & 3;                 // 0..3  (N group)
    const int m0 = blockIdx.y * 64 + wmg * 32;
    const int n0 = blockIdx.x * 256 + wng * 64;
    if (n0 >= N) return;                      // wave-uniform

    const int nr   = lane & 15;               // row-in-tile (A) / col-in-tile (B,C)
    const int half = lane >> 4;

    v8f c00 = {}, c01 = {}, c02 = {}, c03 = {};
    v8f c10 = {}, c11 = {}, c12 = {}, c13 = {};

    // A fragments: lane holds row m; K chunk [k0 + half*16, +16)
    const _Float16* arow0 = A + (size_t)(m0 +      nr) * lda;
    const _Float16* arow1 = A + (size_t)(m0 + 16 + nr) * lda;
    // B fragments: lane holds col n = nr, K chunk [k0 + half*16, +16)
    const _Float16* b0p = W + (size_t)(n0 +  0 + nr) * ldb + half * 16;
    const _Float16* b1p = W + (size_t)(n0 + 16 + nr) * ldb + half * 16;
    const _Float16* b2p = W + (size_t)(n0 + 32 + nr) * ldb + half * 16;
    const _Float16* b3p = W + (size_t)(n0 + 48 + nr) * ldb + half * 16;

    for (int k0 = 0; k0 < K; k0 += 32) {
        v8h t0, t1;
        t0 = *(const v8h*)(arow0 + k0 + half * 16);
        t1 = *(const v8h*)(arow0 + k0 + half * 16 + 8);
        v16h a0 = __builtin_shufflevector(t0, t1, 0,1,2,3,4,5,6,7,8,9,10,11,12,13,14,15);
        t0 = *(const v8h*)(arow1 + k0 + half * 16);
        t1 = *(const v8h*)(arow1 + k0 + half * 16 + 8);
        v16h a1 = __builtin_shufflevector(t0, t1, 0,1,2,3,4,5,6,7,8,9,10,11,12,13,14,15);

        t0 = *(const v8h*)(b0p + k0);  t1 = *(const v8h*)(b0p + k0 + 8);
        v16h b0 = __builtin_shufflevector(t0, t1, 0,1,2,3,4,5,6,7,8,9,10,11,12,13,14,15);
        t0 = *(const v8h*)(b1p + k0);  t1 = *(const v8h*)(b1p + k0 + 8);
        v16h b1 = __builtin_shufflevector(t0, t1, 0,1,2,3,4,5,6,7,8,9,10,11,12,13,14,15);
        t0 = *(const v8h*)(b2p + k0);  t1 = *(const v8h*)(b2p + k0 + 8);
        v16h b2 = __builtin_shufflevector(t0, t1, 0,1,2,3,4,5,6,7,8,9,10,11,12,13,14,15);
        t0 = *(const v8h*)(b3p + k0);  t1 = *(const v8h*)(b3p + k0 + 8);
        v16h b3 = __builtin_shufflevector(t0, t1, 0,1,2,3,4,5,6,7,8,9,10,11,12,13,14,15);

        c00 = __builtin_amdgcn_wmma_f32_16x16x32_f16(false, a0, false, b0, (short)0, c00, false, false);
        c01 = __builtin_amdgcn_wmma_f32_16x16x32_f16(false, a0, false, b1, (short)0, c01, false, false);
        c02 = __builtin_amdgcn_wmma_f32_16x16x32_f16(false, a0, false, b2, (short)0, c02, false, false);
        c03 = __builtin_amdgcn_wmma_f32_16x16x32_f16(false, a0, false, b3, (short)0, c03, false, false);
        c10 = __builtin_amdgcn_wmma_f32_16x16x32_f16(false, a1, false, b0, (short)0, c10, false, false);
        c11 = __builtin_amdgcn_wmma_f32_16x16x32_f16(false, a1, false, b1, (short)0, c11, false, false);
        c12 = __builtin_amdgcn_wmma_f32_16x16x32_f16(false, a1, false, b2, (short)0, c12, false, false);
        c13 = __builtin_amdgcn_wmma_f32_16x16x32_f16(false, a1, false, b3, (short)0, c13, false, false);
    }

    float bv0 = 0.f, bv1 = 0.f, bv2 = 0.f, bv3 = 0.f;
    if (bias) {
        bv0 = bias[n0 +  0 + nr];
        bv1 = bias[n0 + 16 + nr];
        bv2 = bias[n0 + 32 + nr];
        bv3 = bias[n0 + 48 + nr];
    }

    // C layout: lane L -> col n = L%16, rows mtile + (L/16)*8 + r (r = VGPR idx)
    #pragma unroll
    for (int r = 0; r < 8; ++r) {
        size_t row0 = (size_t)(m0 +      half * 8 + r) * ldc;
        size_t row1 = (size_t)(m0 + 16 + half * 8 + r) * ldc;
        float v00 = c00[r] + bv0, v01 = c01[r] + bv1, v02 = c02[r] + bv2, v03 = c03[r] + bv3;
        float v10 = c10[r] + bv0, v11 = c11[r] + bv1, v12 = c12[r] + bv2, v13 = c13[r] + bv3;
        if (nt_store) {   // streaming store for the 524 MB logits (protect L2)
            __builtin_nontemporal_store(v00, &C[row0 + n0 +  0 + nr]);
            __builtin_nontemporal_store(v01, &C[row0 + n0 + 16 + nr]);
            __builtin_nontemporal_store(v02, &C[row0 + n0 + 32 + nr]);
            __builtin_nontemporal_store(v03, &C[row0 + n0 + 48 + nr]);
            __builtin_nontemporal_store(v10, &C[row1 + n0 +  0 + nr]);
            __builtin_nontemporal_store(v11, &C[row1 + n0 + 16 + nr]);
            __builtin_nontemporal_store(v12, &C[row1 + n0 + 32 + nr]);
            __builtin_nontemporal_store(v13, &C[row1 + n0 + 48 + nr]);
        } else {
            C[row0 + n0 +  0 + nr] = v00;
            C[row0 + n0 + 16 + nr] = v01;
            C[row0 + n0 + 32 + nr] = v02;
            C[row0 + n0 + 48 + nr] = v03;
            C[row1 + n0 +  0 + nr] = v10;
            C[row1 + n0 + 16 + nr] = v11;
            C[row1 + n0 + 32 + nr] = v12;
            C[row1 + n0 + 48 + nr] = v13;
        }
    }
}

// ---------------------------------------------------------------------------
// GRU gate update for one timestep (fp32, biases folded in here)
// ---------------------------------------------------------------------------
__global__ void k_gru_gate(const float* __restrict__ xp, const float* __restrict__ gh,
                           const float* __restrict__ b_ih, const float* __restrict__ b_hh,
                           const int* __restrict__ lengths, int t,
                           float* __restrict__ h, _Float16* __restrict__ h16,
                           _Float16* __restrict__ hx16, float* __restrict__ hx_out) {
    int idx = blockIdx.x * blockDim.x + threadIdx.x;
    if (idx >= BATCH * HID_P) return;
    int b = idx / HID_P, j = idx % HID_P;
    size_t hxrow = (size_t)b * TLEN + t;
    if (j >= HID) {                    // keep f16 pads zeroed
        hx16[hxrow * HID_P + j] = (_Float16)0.f;
        return;
    }
    size_t xrow = hxrow * GATES_P;
    size_t grow = (size_t)b * GATES_P;
    float xr = xp[xrow + j]           + b_ih[j];
    float xz = xp[xrow + HID + j]     + b_ih[HID + j];
    float xn = xp[xrow + 2 * HID + j] + b_ih[2 * HID + j];
    float gr = gh[grow + j]           + b_hh[j];
    float gz = gh[grow + HID + j]     + b_hh[HID + j];
    float gn = gh[grow + 2 * HID + j] + b_hh[2 * HID + j];

    float r  = 1.f / (1.f + __expf(-(xr + gr)));
    float zg = 1.f / (1.f + __expf(-(xz + gz)));
    float nn = tanhf(xn + r * gn);

    float hp = h[(size_t)b * HID_P + j];
    float hn = (1.f - zg) * nn + zg * hp;
    bool  m  = t < lengths[b];
    float hc = m ? hn : hp;

    h[(size_t)b * HID_P + j]   = hc;
    h16[(size_t)b * HID_P + j] = (_Float16)hc;

    float hx = m ? hn : 0.f;
    hx_out[hxrow * HID + j]   = hx;              // fp32 output #1 (B,T,HID)
    hx16[hxrow * HID_P + j]   = (_Float16)hx;    // f16 A-matrix for logits GEMM
}

// ---------------------------------------------------------------------------
extern "C" void kernel_launch(void* const* d_in, const int* in_sizes, int n_in,
                              void* d_out, int out_size, void* d_ws, size_t ws_size,
                              hipStream_t stream) {
    const float* z        = (const float*)d_in[0];
    const int*   l        = (const int*)  d_in[1];
    const int*   x        = (const int*)  d_in[2];
    const int*   lengths  = (const int*)  d_in[3];
    const float* emb      = (const float*)d_in[4];
    const float* attr_emb = (const float*)d_in[5];
    const float* W_ih     = (const float*)d_in[6];
    const float* W_hh     = (const float*)d_in[7];
    const float* b_ih     = (const float*)d_in[8];
    const float* b_hh     = (const float*)d_in[9];
    const float* W_out    = (const float*)d_in[10];
    const float* b_out    = (const float*)d_in[11];

    float* out    = (float*)d_out;
    float* hx_out = out;                                   // B*T*HID fp32
    float* logits = out + (size_t)ROWS * HID;              // B*T*VOCAB fp32

    // workspace carve-up (256B aligned blocks), ~93 MB total
    char* ws = (char*)d_ws;
    size_t off = 0;
    auto take = [&](size_t bytes) {
        size_t o = off;
        off += (bytes + 255) & ~(size_t)255;
        return (void*)(ws + o);
    };
    float*    h_f32  = (float*)   take((size_t)BATCH * HID_P * 4);
    float*    gh     = (float*)   take((size_t)BATCH * GATES_P * 4);
    float*    xp     = (float*)   take((size_t)ROWS * GATES_P * 4);
    _Float16* h16    = (_Float16*)take((size_t)BATCH * HID_P * 2);
    _Float16* hx16   = (_Float16*)take((size_t)ROWS * HID_P * 2);
    _Float16* xe     = (_Float16*)take((size_t)ROWS * EMB_P * 2);
    _Float16* Wih16  = (_Float16*)take((size_t)GATES_P * EMB_P * 2);
    _Float16* Whh16  = (_Float16*)take((size_t)GATES_P * HID_P * 2);
    _Float16* Wout16 = (_Float16*)take((size_t)VOCAB * HID_P * 2);

    const int TB = 256;
    auto blocks = [](long long n) { return (unsigned)((n + 255) / 256); };

    // ---- setup / conversion --------------------------------------------
    k_prep_h0<<<blocks(BATCH * HID_P), TB, 0, stream>>>(z, l, attr_emb, h_f32, h16);
    k_cvt_pad<<<blocks((long long)GATES_P * EMB_P), TB, 0, stream>>>(W_ih, Wih16, GATES, EMB, GATES_P, EMB_P);
    k_cvt_pad<<<blocks((long long)GATES_P * HID_P), TB, 0, stream>>>(W_hh, Whh16, GATES, HID, GATES_P, HID_P);
    k_cvt_pad<<<blocks((long long)VOCAB * HID_P), TB, 0, stream>>>(W_out, Wout16, VOCAB, HID, VOCAB, HID_P);
    k_gather_emb<<<blocks((long long)ROWS * EMB_P), TB, 0, stream>>>(x, emb, xe);

    // ---- xp = xe @ W_ih^T : (4096 x 320) x (2112 x 320)^T ---------------
    {
        dim3 grid((GATES_P + 255) / 256, ROWS / 64);
        k_wmma_gemm<<<grid, TB, 0, stream>>>(xe, Wih16, xp, nullptr,
                                             GATES_P, EMB_P, EMB_P, EMB_P, GATES_P, 0);
    }

    // ---- sequential scan over T ----------------------------------------
    {
        dim3 ggrid((GATES_P + 255) / 256, BATCH / 64);
        for (int t = 0; t < TLEN; ++t) {
            k_wmma_gemm<<<ggrid, TB, 0, stream>>>(h16, Whh16, gh, nullptr,
                                                  GATES_P, HID_P, HID_P, HID_P, GATES_P, 0);
            k_gru_gate<<<blocks(BATCH * HID_P), TB, 0, stream>>>(
                xp, gh, b_ih, b_hh, lengths, t, h_f32, h16, hx16, hx_out);
        }
    }

    // ---- logits = hx @ W_out^T + b_out : (4096 x 704) x (32000 x 704)^T -
    {
        dim3 grid(VOCAB / 256, ROWS / 64);
        k_wmma_gemm<<<grid, TB, 0, stream>>>(hx16, Wout16, logits, b_out,
                                             VOCAB, HID_P, HID_P, HID_P, VOCAB, 1);
    }
}
